// AlignedMPNN_57913339019279
// MI455X (gfx1250) — matrix operand order repaired and checked by reading, not compiled, not run on hardware
//
#include <hip/hip_runtime.h>
#include <hip/hip_bf16.h>

// Dims from the reference
#define Bx 8
#define Nx 256
#define Fx 128
#define Dx 128

typedef __attribute__((ext_vector_type(16))) _Float16 v16h;
typedef __attribute__((ext_vector_type(8)))  float    v8f;

// ws layout (floats)
#define MSG1_OFF 0
#define MSG2_OFF (Bx * 257 * Dx)          // 263168
#define H1_OFF   (2 * Bx * 257 * Dx)
#define MSGS_OFF (3 * Bx * 257 * Dx)      // [B,256,D] combined msg1+colmax

// ---------------- helpers ----------------

// A fragment: 16x32 f16. lane: M = lane%16, g = lane/16.
// element e in [0,16): K = kb + g*16 + e
// src = per-lane fp32 row pointer covering this 128-wide K half (nullptr => zero row)
__device__ __forceinline__ void load_a(const float* src, int kb, int g,
                                       v16h& ahi, v16h& alo) {
  if (src) {
    const float4* q0 = (const float4*)(src + kb + g * 16);
    float4 x0 = q0[0];
    float4 x1 = q0[1];
    const float4* q1 = (const float4*)(src + kb + g * 16 + 8);
    float4 y0 = q1[0];
    float4 y1 = q1[1];
    float v[16] = {x0.x, x0.y, x0.z, x0.w, x1.x, x1.y, x1.z, x1.w,
                   y0.x, y0.y, y0.z, y0.w, y1.x, y1.y, y1.z, y1.w};
#pragma unroll
    for (int e = 0; e < 16; ++e) {
      _Float16 hh = (_Float16)v[e];
      ahi[e] = hh;
      alo[e] = (_Float16)(v[e] - (float)hh);
    }
  } else {
#pragma unroll
    for (int e = 0; e < 16; ++e) { ahi[e] = (_Float16)0.f; alo[e] = (_Float16)0.f; }
  }
}

// B fragment: 32x16 f16. lane: N = lane%16, g = lane/16; element h: K = k0 + h
// LDS holds W^T as halves: [d][k], row pitch 264 halves (528B) to spread banks.
__device__ __forceinline__ v16h load_b_lds(const _Float16* sw, int d, int k0) {
  union { int4 q[2]; v16h v; } u;
  const int4* p = (const int4*)(sw + d * 264 + k0);
  u.q[0] = p[0];
  u.q[1] = p[1];
  return u.v;
}

// B fragment straight from a row-major fp32 weight [K][ldw] in global (L2-hot).
__device__ __forceinline__ void load_b_global(const float* __restrict__ W, int ldw,
                                              int d, int k0, v16h& bhi, v16h& blo) {
#pragma unroll
  for (int h = 0; h < 16; ++h) {
    float w = W[(size_t)(k0 + h) * ldw + d];
    _Float16 hh = (_Float16)w;
    bhi[h] = hh;
    blo[h] = (_Float16)(w - (float)hh);
  }
}

// Split-f16 fp32-ish product: c += ahi*blo + alo*bhi + ahi*bhi  (fp32 accumulate)
__device__ __forceinline__ v8f wmma3(v8f c, v16h ahi, v16h alo, v16h bhi, v16h blo) {
  c = __builtin_amdgcn_wmma_f32_16x16x32_f16(false, ahi, false, blo, (short)0, c, false, false);
  c = __builtin_amdgcn_wmma_f32_16x16x32_f16(false, alo, false, bhi, (short)0, c, false, false);
  c = __builtin_amdgcn_wmma_f32_16x16x32_f16(false, ahi, false, bhi, (short)0, c, false, false);
  return c;
}

// ---------------- kernel 1: et = [edge|e_hid] @ We + be  (the 768MB stream) ----------------
// 1024 blocks x 256 threads; each wave does 4 strips of 16 rows x 128 cols, K=256.
__global__ void __launch_bounds__(256, 1)
et_gemm_kernel(const float* __restrict__ edge, const float* __restrict__ ehid,
               const float* __restrict__ We,   const float* __restrict__ be,
               float* __restrict__ out) {
  extern __shared__ _Float16 smem[];
  _Float16* sHi = smem;                 // 128 rows x 264 halves
  _Float16* sLo = smem + 128 * 264;

  for (int idx = threadIdx.x; idx < 256 * 128; idx += blockDim.x) {
    int k = idx >> 7, d = idx & 127;    // We is [K=256][D=128]
    float w = We[idx];
    _Float16 hh = (_Float16)w;
    sHi[d * 264 + k] = hh;
    sLo[d * 264 + k] = (_Float16)(w - (float)hh);
  }
  __syncthreads();

  const int lane = threadIdx.x & 31;
  const int wave = threadIdx.x >> 5;
  const int n = lane & 15, g = lane >> 4;

  for (int it = 0; it < 4; ++it) {
    const int s = (blockIdx.x * 8 + wave) + it * 8192;   // strip id, 32768 total
    const size_t m0 = (size_t)s * 16;
    const float* p0 = edge + (m0 + n) * 128;
    const float* p1 = ehid + (m0 + n) * 128;

    v8f c[8];
#pragma unroll
    for (int t = 0; t < 8; ++t)
#pragma unroll
      for (int r = 0; r < 8; ++r) c[t][r] = 0.f;

#pragma clang loop unroll(disable)
    for (int ks = 0; ks < 8; ++ks) {
      const int kb = ks * 32;
      v16h ahi, alo;
      load_a((kb < 128) ? p0 : p1, kb & 127, g, ahi, alo);
      __builtin_amdgcn_sched_barrier(0);
      const int k0 = kb + g * 16;
#pragma unroll
      for (int t = 0; t < 8; ++t) {
        v16h bhi = load_b_lds(sHi, t * 16 + n, k0);
        v16h blo = load_b_lds(sLo, t * 16 + n, k0);
        c[t] = wmma3(c[t], ahi, alo, bhi, blo);
        __builtin_amdgcn_sched_barrier(0);
      }
    }

    float* orow = out + m0 * 128;
#pragma unroll
    for (int t = 0; t < 8; ++t) {
      float bv = be[t * 16 + n];
#pragma unroll
      for (int r = 0; r < 8; ++r)
        orow[(size_t)(r + 8 * g) * 128 + t * 16 + n] = c[t][r] + bv;
    }
  }
}

// ---------------- kernel 2: msg1/msg2/h1 = nt @ {W_m1,W_m2,W_o1} + bias ----------------
// nt[b,j,:] = concat(node[b,j],hid[b,j]) for j<256, zeros for j==256 (virtual node).
// 17 blocks x 256 threads = 136 waves = 8 batches * 17 strips of 16 rows.
__global__ void __launch_bounds__(256, 1)
msg_gemm_kernel(const float* __restrict__ node, const float* __restrict__ hid,
                const float* __restrict__ Wm1, const float* __restrict__ bm1,
                const float* __restrict__ Wm2, const float* __restrict__ bm2,
                const float* __restrict__ Wo1, const float* __restrict__ bo1,
                float* __restrict__ ws) {
  const int lane = threadIdx.x & 31, wave = threadIdx.x >> 5;
  const int n = lane & 15, g = lane >> 4;
  const int s = blockIdx.x * 8 + wave;     // 0..135
  const int b = s / 17, ls = s % 17;
  const int j0 = ls * 16;
  const int j = j0 + n;                    // per-lane row in [0,272)
  const float* p0 = nullptr;
  const float* p1 = nullptr;
  if (j < 256) {
    p0 = node + ((size_t)b * 256 + j) * 128;
    p1 = hid  + ((size_t)b * 256 + j) * 128;
  }
  const float* Ws[3]   = {Wm1, Wm2, Wo1};
  const float* bsv[3]  = {bm1, bm2, bo1};
  float*       outs[3] = {ws + MSG1_OFF, ws + MSG2_OFF, ws + H1_OFF};

#pragma unroll
  for (int w = 0; w < 3; ++w) {
    v8f c[8];
#pragma unroll
    for (int t = 0; t < 8; ++t)
#pragma unroll
      for (int r = 0; r < 8; ++r) c[t][r] = 0.f;

#pragma clang loop unroll(disable)
    for (int ks = 0; ks < 8; ++ks) {
      const int kb = ks * 32;
      v16h ahi, alo;
      load_a((kb < 128) ? p0 : p1, kb & 127, g, ahi, alo);
      __builtin_amdgcn_sched_barrier(0);
#pragma unroll
      for (int t = 0; t < 8; ++t) {
        v16h bhi, blo;
        load_b_global(Ws[w], 128, t * 16 + n, kb + g * 16, bhi, blo);
        c[t] = wmma3(c[t], ahi, alo, bhi, blo);
        __builtin_amdgcn_sched_barrier(0);
      }
    }
#pragma unroll
    for (int t = 0; t < 8; ++t) {
      float bv = bsv[w][t * 16 + n];
#pragma unroll
      for (int r = 0; r < 8; ++r) {
        int jr = j0 + r + 8 * g;
        if (jr <= 256)
          outs[w][((size_t)b * 257 + jr) * 128 + t * 16 + n] = c[t][r] + bv;
      }
    }
  }
}

// ---------------- kernel 3: masked column max + combine ----------------
// msgs[b,j,d] = msg1[b,j,d] + max(msg2[b,256,d], max_{i<256, adj[b,i,j]!=0} msg2[b,i,d])
__global__ void colmax_kernel(const int* __restrict__ adj, const float* __restrict__ ws,
                              float* __restrict__ msgs_out) {
  const int bj = blockIdx.x;        // 0..2047
  const int b = bj >> 8, j = bj & 255;
  const int d = threadIdx.x;        // 0..127
  const float* msg1 = ws + MSG1_OFF;
  const float* msg2 = ws + MSG2_OFF;

  float m = msg2[((size_t)b * 257 + 256) * 128 + d];   // virtual row always connected
  const int*   acol = adj + (size_t)b * 256 * 256 + j; // adj[b][i][j], stride 256
  const float* m2b  = msg2 + (size_t)b * 257 * 128 + d;
  for (int i = 0; i < 256; ++i) {
    float v = m2b[(size_t)i * 128];
    if (acol[(size_t)i * 256] != 0) m = fmaxf(m, v);
  }
  msgs_out[((size_t)b * 256 + j) * 128 + d] =
      msg1[((size_t)b * 257 + j) * 128 + d] + m;
}

// ---------------- kernel 4: ret = h1 + b_o2 + msgs @ W_o2 (rows j<256 only) ----------------
// 16 blocks x 256 threads = 128 strips of 16 rows, K=128.
__global__ void __launch_bounds__(256, 1)
out_gemm_kernel(const float* __restrict__ ws, const float* __restrict__ Wo2,
                const float* __restrict__ bo2, float* __restrict__ ret) {
  const float* h1   = ws + H1_OFF;
  const float* msgs = ws + MSGS_OFF;
  const int lane = threadIdx.x & 31, wave = threadIdx.x >> 5;
  const int n = lane & 15, g = lane >> 4;
  const int s = blockIdx.x * 8 + wave;   // 0..127
  const int m0 = s * 16;                 // global row = b*256+j; strips never straddle b
  const int b = m0 >> 8;
  const float* p0 = msgs + ((size_t)m0 + n) * 128;

  v8f c[8];
#pragma unroll
  for (int t = 0; t < 8; ++t)
#pragma unroll
    for (int r = 0; r < 8; ++r) c[t][r] = 0.f;

#pragma clang loop unroll(disable)
  for (int ks = 0; ks < 4; ++ks) {
    const int kb = ks * 32;
    v16h ahi, alo;
    load_a(p0, kb, g, ahi, alo);
    __builtin_amdgcn_sched_barrier(0);
#pragma unroll
    for (int t = 0; t < 8; ++t) {
      v16h bhi, blo;
      load_b_global(Wo2, 128, t * 16 + n, kb + g * 16, bhi, blo);
      c[t] = wmma3(c[t], ahi, alo, bhi, blo);
      __builtin_amdgcn_sched_barrier(0);
    }
  }
#pragma unroll
  for (int t = 0; t < 8; ++t) {
    float bv = bo2[t * 16 + n];
#pragma unroll
    for (int r = 0; r < 8; ++r) {
      int j = (m0 + r + 8 * g) & 255;
      ret[((size_t)m0 + r + 8 * g) * 128 + t * 16 + n] =
          c[t][r] + bv + h1[((size_t)b * 257 + j) * 128 + t * 16 + n];
    }
  }
}

// ---------------- launch ----------------
extern "C" void kernel_launch(void* const* d_in, const int* in_sizes, int n_in,
                              void* d_out, int out_size, void* d_ws, size_t ws_size,
                              hipStream_t stream) {
  (void)in_sizes; (void)n_in; (void)out_size; (void)ws_size;
  const float* node = (const float*)d_in[0];
  const float* edge = (const float*)d_in[1];
  // d_in[2] = graph_fts (unused by the reference)
  const int*   adj  = (const int*)d_in[3];
  const float* hid  = (const float*)d_in[4];
  const float* ehid = (const float*)d_in[5];
  const float* We   = (const float*)d_in[6];
  const float* be   = (const float*)d_in[7];
  const float* Wm1  = (const float*)d_in[8];
  const float* bm1  = (const float*)d_in[9];
  const float* Wm2  = (const float*)d_in[10];
  const float* bm2  = (const float*)d_in[11];
  const float* Wo1  = (const float*)d_in[12];
  const float* bo1  = (const float*)d_in[13];
  const float* Wo2  = (const float*)d_in[14];
  const float* bo2  = (const float*)d_in[15];

  float* ret = (float*)d_out;                  // [B,N,D]
  float* et  = ret + (size_t)Bx * Nx * Dx;     // [B,N,N,D]
  float* ws  = (float*)d_ws;

  const size_t smem = 2u * 128u * 264u * sizeof(_Float16);  // 132KB of the 320KB WGP LDS
  (void)hipFuncSetAttribute((const void*)et_gemm_kernel,
                            hipFuncAttributeMaxDynamicSharedMemorySize, (int)smem);

  et_gemm_kernel<<<1024, 256, smem, stream>>>(edge, ehid, We, be, et);
  msg_gemm_kernel<<<17, 256, 0, stream>>>(node, hid, Wm1, bm1, Wm2, bm2, Wo1, bo1, ws);
  colmax_kernel<<<2048, 128, 0, stream>>>(adj, ws, ws + MSGS_OFF);
  out_gemm_kernel<<<16, 256, 0, stream>>>(ws, Wo2, bo2, ret);
}
